// CameraHead_69509750718791
// MI455X (gfx1250) — compile-verified, matrix operand order checked by reading, not run on hardware
//
#include <hip/hip_runtime.h>
#include <math.h>

typedef unsigned short u16;
typedef unsigned int   u32;
typedef __attribute__((ext_vector_type(16))) __bf16 v16bf;
typedef __attribute__((ext_vector_type(8)))  float  v8f;

#define S_TOK 1024
#define CDIM  2048
#define HEADS 16
#define HDIM  128
#define C3    6144
#define HID   8192

// ---- CDNA5 async global->LDS (ASYNCcnt-tracked, bypasses VGPRs) ----
#if defined(__has_builtin)
#if __has_builtin(__builtin_amdgcn_global_load_async_to_lds_b128)
#define HAVE_ASYNC_LDS 1
#endif
#endif
#ifndef HAVE_ASYNC_LDS
#define HAVE_ASYNC_LDS 0
#endif

typedef int v4i_g __attribute__((vector_size(16)));   // GCC-style int4 (matches builtin signature)
typedef __attribute__((address_space(1))) v4i_g* gptr_v4i;
typedef __attribute__((address_space(3))) v4i_g* lptr_v4i;

__device__ __forceinline__ void async_copy16(const u16* g, u32* lds) {
#if HAVE_ASYNC_LDS
    __builtin_amdgcn_global_load_async_to_lds_b128(
        (gptr_v4i)(uintptr_t)g,
        (lptr_v4i)(uintptr_t)(u32)(uintptr_t)lds,
        0, 0);
#else
    *(uint4*)lds = *(const uint4*)g;
#endif
}
__device__ __forceinline__ void wait_async() {
#if HAVE_ASYNC_LDS
#if __has_builtin(__builtin_amdgcn_s_wait_asynccnt)
    __builtin_amdgcn_s_wait_asynccnt(0);
#else
    asm volatile("s_wait_asynccnt 0x0" ::: "memory");
#endif
#endif
}

// ---------------- helpers ----------------
__device__ __forceinline__ u16 f2b(float x) {
    u32 u = __float_as_uint(x);
    u32 r = (u + 0x7FFFu + ((u >> 16) & 1u)) >> 16;
    return (u16)r;
}
__device__ __forceinline__ float b2f(u16 v) { return __uint_as_float(((u32)v) << 16); }
__device__ __forceinline__ float gelu_exact(float x) {
    return 0.5f * x * (1.0f + erff(x * 0.70710678118654752f));
}

// ---------------- fp32 -> bf16 weight conversion ----------------
__global__ void k_f32_to_bf16(const float* __restrict__ src, u16* __restrict__ dst, int n) {
    int i = blockIdx.x * 256 + threadIdx.x;
    if (i < n) dst[i] = f2b(src[i]);
}

// ---------------- LayerNorm (one block per row) ----------------
__global__ void k_layernorm(const float* __restrict__ in, int ld,
                            const float* __restrict__ g, const float* __restrict__ b,
                            float* __restrict__ outf, u16* __restrict__ outb,
                            int C, float eps) {
    int row = blockIdx.x;
    const float* r = in + (size_t)row * ld;
    float s = 0.f, s2 = 0.f;
    for (int c = threadIdx.x; c < C; c += 256) { float v = r[c]; s += v; s2 += v * v; }
    #pragma unroll
    for (int off = 16; off > 0; off >>= 1) { s += __shfl_down(s, off, 32); s2 += __shfl_down(s2, off, 32); }
    __shared__ float rs[8], rs2[8];
    int lane = threadIdx.x & 31, wid = threadIdx.x >> 5;
    if (lane == 0) { rs[wid] = s; rs2[wid] = s2; }
    __syncthreads();
    if (threadIdx.x == 0) {
        float a = 0.f, a2 = 0.f;
        #pragma unroll
        for (int w = 0; w < 8; ++w) { a += rs[w]; a2 += rs2[w]; }
        rs[0] = a; rs2[0] = a2;
    }
    __syncthreads();
    float mean = rs[0] / C;
    float var  = rs2[0] / C - mean * mean;
    float inv  = rsqrtf(var + eps);
    for (int c = threadIdx.x; c < C; c += 256) {
        float v = (r[c] - mean) * inv;
        if (g) v = v * g[c] + b[c];
        if (outf) outf[(size_t)row * C + c] = v;
        if (outb) outb[(size_t)row * C + c] = f2b(v);
    }
}

// ---------------- WMMA bf16 GEMM (async double-buffered LDS) ----------------
enum { EPI_BF16 = 0, EPI_BF16_GELU = 1, EPI_BF16_SILU = 2, EPI_F32 = 3, EPI_F32_RES = 4 };

union FragAB { v16bf v; u32 u[8]; };

template <int TRANSB>
__global__ __launch_bounds__(256)
void k_gemm(const u16* __restrict__ A, int lda, long long sAz,
            const u16* __restrict__ B, int ldb, long long sBz,
            float* __restrict__ Cf, u16* __restrict__ Cb, int ldc, long long sCz,
            const float* __restrict__ bias, const float* __restrict__ colscale,
            int M, int N, int K, float alpha, int epi, int causal) {
    if (causal && (int)blockIdx.x > (int)blockIdx.y) return;  // tile fully above diagonal
    __shared__ u32 As[2][2048];  // 128 rows x 32 bf16, double-buffered
    __shared__ u32 Bs[2][2048];  // TRANSB: 128x32 ; else 32x128 (u16 view)
    const int z = blockIdx.z;
    const u16* Ab = A + (size_t)z * (size_t)sAz;
    const u16* Bb = B + (size_t)z * (size_t)sBz;
    const size_t cOff = (size_t)z * (size_t)sCz;
    const int mb = blockIdx.y, nb = blockIdx.x;
    const int tid = threadIdx.x, lane = tid & 31, wid = tid >> 5;
    const int wm = wid & 3, wn = wid >> 2;           // 4x2 wave grid
    const int m0 = wm * 32, n0 = wn * 64;            // wave tile: 32 x 64
    const int lm  = lane & 15;
    const int kbu = (lane >> 4) * 4;                 // uint col base (K half select)

    // ---- per-thread staging slots (2 x 16B for A, 2 x 16B for B) ----
    const int ia0 = tid, ia1 = tid + 256;
    const int ra0 = ia0 >> 2, qa0 = ia0 & 3, ra1 = ia1 >> 2, qa1 = ia1 & 3;
    const u16* gA0 = Ab + (size_t)(mb * 128 + ra0) * lda + qa0 * 8;
    const u16* gA1 = Ab + (size_t)(mb * 128 + ra1) * lda + qa1 * 8;
    const u32 la0 = ra0 * 16 + qa0 * 4, la1 = ra1 * 16 + qa1 * 4;
    const u16 *gB0, *gB1;
    u32 lb0, lb1;
    size_t bStep;
    if (TRANSB) {
        gB0 = Bb + (size_t)(nb * 128 + ra0) * ldb + qa0 * 8;
        gB1 = Bb + (size_t)(nb * 128 + ra1) * ldb + qa1 * 8;
        lb0 = la0; lb1 = la1;
        bStep = 32;
    } else {
        const int rb0 = ia0 >> 4, qb0 = ia0 & 15, rb1 = ia1 >> 4, qb1 = ia1 & 15;
        gB0 = Bb + (size_t)rb0 * ldb + nb * 128 + qb0 * 8;
        gB1 = Bb + (size_t)rb1 * ldb + nb * 128 + qb1 * 8;
        lb0 = rb0 * 64 + qb0 * 4; lb1 = rb1 * 64 + qb1 * 4;
        bStep = (size_t)32 * ldb;
    }

    v8f acc[2][4];
    #pragma unroll
    for (int i = 0; i < 2; ++i)
        #pragma unroll
        for (int j = 0; j < 4; ++j)
            acc[i][j] = (v8f){0.f, 0.f, 0.f, 0.f, 0.f, 0.f, 0.f, 0.f};

    const int ktiles = K >> 5;
    auto stage = [&](int kt, int p) {
        const size_t ka = (size_t)kt * 32;
        const size_t kb = (size_t)kt * bStep;
        async_copy16(gA0 + ka, &As[p][la0]);
        async_copy16(gA1 + ka, &As[p][la1]);
        async_copy16(gB0 + kb, &Bs[p][lb0]);
        async_copy16(gB1 + kb, &Bs[p][lb1]);
    };

    stage(0, 0);
    wait_async();
    __syncthreads();

    for (int kt = 0; kt < ktiles; ++kt) {
        const int p = kt & 1;
        if (kt + 1 < ktiles) stage(kt + 1, p ^ 1);        // stream next tile into other buffer
        if (kt + 2 < ktiles) {                            // L2 prefetch of the weight stream
            __builtin_prefetch(gB0 + (size_t)(kt + 2) * bStep, 0, 1);
            __builtin_prefetch(gB1 + (size_t)(kt + 2) * bStep, 0, 1);
        }

        // ---- fragments per ISA 16-bit A/B lane layout ----
        FragAB fa[2];
        #pragma unroll
        for (int i = 0; i < 2; ++i) {
            const int r = m0 + i * 16 + lm;
            #pragma unroll
            for (int pp = 0; pp < 8; ++pp) {
                const int cu = (pp < 4) ? (kbu + pp) : (8 + kbu + (pp - 4));
                fa[i].u[pp] = As[p][r * 16 + cu];
            }
        }
        FragAB fb[4];
        if (TRANSB) {
            #pragma unroll
            for (int j = 0; j < 4; ++j) {
                const int r = n0 + j * 16 + lm;
                #pragma unroll
                for (int pp = 0; pp < 8; ++pp) {
                    const int cu = (pp < 4) ? (kbu + pp) : (8 + kbu + (pp - 4));
                    fb[j].u[pp] = Bs[p][r * 16 + cu];
                }
            }
        } else {
            const u16* Bh = (const u16*)&Bs[p][0];
            #pragma unroll
            for (int j = 0; j < 4; ++j) {
                const int n = n0 + j * 16 + lm;
                #pragma unroll
                for (int pp = 0; pp < 8; ++pp) {
                    const int cu = (pp < 4) ? (kbu + pp) : (8 + kbu + (pp - 4));
                    const int k0 = cu * 2;
                    u32 lo = Bh[k0 * 128 + n];
                    u32 hi = Bh[(k0 + 1) * 128 + n];
                    fb[j].u[pp] = lo | (hi << 16);
                }
            }
        }
        #pragma unroll
        for (int i = 0; i < 2; ++i)
            #pragma unroll
            for (int j = 0; j < 4; ++j)
                acc[i][j] = __builtin_amdgcn_wmma_f32_16x16x32_bf16(
                    false, fa[i].v, false, fb[j].v, (short)0, acc[i][j], false, false);

        wait_async();       // our async stores into buffer p^1 done
        __syncthreads();    // all waves: done reading p, done filling p^1
    }

    // ---- epilogue (C layout: VGPR vg -> M = vg + 8*(lane>=16), N = lane&15) ----
    const int rsel = (lane >> 4) * 8;
    #pragma unroll
    for (int i = 0; i < 2; ++i) {
        #pragma unroll
        for (int j = 0; j < 4; ++j) {
            const int gcol = nb * 128 + n0 + j * 16 + lm;
            const float bv = bias ? bias[gcol] : 0.0f;
            #pragma unroll
            for (int vg = 0; vg < 8; ++vg) {
                const int grow = mb * 128 + m0 + i * 16 + rsel + vg;
                float v = acc[i][j][vg] * alpha + bv;
                const size_t o = cOff + (size_t)grow * ldc + gcol;
                if (epi == EPI_BF16)           Cb[o] = f2b(v);
                else if (epi == EPI_BF16_GELU) Cb[o] = f2b(gelu_exact(v));
                else if (epi == EPI_BF16_SILU) Cb[o] = f2b(v / (1.0f + __expf(-v)));
                else if (epi == EPI_F32)       Cf[o] = v;
                else                           Cf[o] = Cf[o] + colscale[gcol] * v;  // x += ls*(acc+bias)
            }
        }
    }
}

// ---------------- causal softmax: one block per (row, head) ----------------
__global__ void k_softmax_causal(const float* __restrict__ scores, u16* __restrict__ probs, int S) {
    const int s = blockIdx.x, h = blockIdx.y;
    const float* r = scores + ((size_t)h * S + s) * S;
    u16* pr = probs + ((size_t)h * S + s) * S;
    const int valid = s + 1;
    __shared__ float red[8];
    int lane = threadIdx.x & 31, wid = threadIdx.x >> 5;

    float mx = -3.4e38f;
    for (int c = threadIdx.x; c < valid; c += 256) mx = fmaxf(mx, r[c]);
    #pragma unroll
    for (int off = 16; off > 0; off >>= 1) mx = fmaxf(mx, __shfl_down(mx, off, 32));
    if (lane == 0) red[wid] = mx;
    __syncthreads();
    if (threadIdx.x == 0) {
        float m = red[0];
        #pragma unroll
        for (int w = 1; w < 8; ++w) m = fmaxf(m, red[w]);
        red[0] = m;
    }
    __syncthreads();
    mx = red[0];
    __syncthreads();

    float sum = 0.f;
    for (int c = threadIdx.x; c < valid; c += 256) sum += __expf(r[c] - mx);
    #pragma unroll
    for (int off = 16; off > 0; off >>= 1) sum += __shfl_down(sum, off, 32);
    if (lane == 0) red[wid] = sum;
    __syncthreads();
    if (threadIdx.x == 0) {
        float a = 0.f;
        #pragma unroll
        for (int w = 0; w < 8; ++w) a += red[w];
        red[0] = a;
    }
    __syncthreads();
    const float inv = 1.0f / red[0];
    for (int c = threadIdx.x; c < S; c += 256)
        pr[c] = (c < valid) ? f2b(__expf(r[c] - mx) * inv) : (u16)0;
}

// ---------------- embed (K=9) + SiLU -> bf16 ----------------
__global__ void k_embed_silu(const float* __restrict__ inp, int inpStride,
                             const float* __restrict__ w, const float* __restrict__ b,
                             u16* __restrict__ out, int C) {
    const int c = blockIdx.x * 256 + threadIdx.x;
    const int s = blockIdx.y;
    const float* ip = inp + (size_t)s * inpStride;
    float acc = b[c];
    #pragma unroll
    for (int j = 0; j < 9; ++j) acc += ip[j] * w[j * C + c];
    out[(size_t)s * C + c] = f2b(acc / (1.0f + __expf(-acc)));
}

// ---------------- modulation: x = gate*(lnp*(1+scale)+shift) + pose ----------------
__global__ void k_modulate(const float* __restrict__ pose, const float* __restrict__ lnp,
                           const float* __restrict__ mod, float* __restrict__ x, int C) {
    const size_t i = (size_t)blockIdx.x * 256 + threadIdx.x;
    const size_t s = i / C, c = i % C;
    const float sh = mod[s * 3 * C + c];
    const float sc = mod[s * 3 * C + C + c];
    const float gt = mod[s * 3 * C + 2 * C + c];
    x[i] = gt * (lnp[i] * (1.0f + sc) + sh) + pose[i];
}

// ---------------- pb2: (S,1024) bf16 x (1024,9) fp32 ----------------
__global__ void k_pb2(const u16* __restrict__ a, const float* __restrict__ w,
                      const float* __restrict__ b, float* __restrict__ delta) {
    const int s = blockIdx.x;
    float acc[9];
    #pragma unroll
    for (int j = 0; j < 9; ++j) acc[j] = 0.f;
    for (int k = threadIdx.x; k < 1024; k += 256) {
        const float av = b2f(a[(size_t)s * 1024 + k]);
        #pragma unroll
        for (int j = 0; j < 9; ++j) acc[j] += av * w[k * 9 + j];
    }
    __shared__ float red[8][9];
    int lane = threadIdx.x & 31, wid = threadIdx.x >> 5;
    #pragma unroll
    for (int j = 0; j < 9; ++j) {
        float v = acc[j];
        #pragma unroll
        for (int off = 16; off > 0; off >>= 1) v += __shfl_down(v, off, 32);
        if (lane == 0) red[wid][j] = v;
    }
    __syncthreads();
    if (threadIdx.x < 9) {
        float t = b[threadIdx.x];
        #pragma unroll
        for (int w8 = 0; w8 < 8; ++w8) t += red[w8][threadIdx.x];
        delta[(size_t)s * 9 + threadIdx.x] = t;
    }
}

// ---------------- pred update + activation + output ----------------
__global__ void k_pred_update(float* __restrict__ pred, const float* __restrict__ delta,
                              float* __restrict__ out, int n, int first) {
    int i = blockIdx.x * 256 + threadIdx.x;
    if (i < n) {
        float p = first ? delta[i] : (pred[i] + delta[i]);
        pred[i] = p;
        int j = i % 9;
        out[i] = (j < 7) ? p : fmaxf(p, 0.f);
    }
}

// ================= host orchestration =================
static inline void gemm(hipStream_t st, bool transB,
                        const u16* A, int lda, long long sAz,
                        const u16* B, int ldb, long long sBz,
                        float* Cf, u16* Cb, int ldc, long long sCz,
                        const float* bias, const float* cs,
                        int M, int N, int K, float alpha, int epi, int causal, int Z) {
    dim3 g(N / 128, M / 128, Z), blk(256);
    if (transB)
        k_gemm<1><<<g, blk, 0, st>>>(A, lda, sAz, B, ldb, sBz, Cf, Cb, ldc, sCz, bias, cs, M, N, K, alpha, epi, causal);
    else
        k_gemm<0><<<g, blk, 0, st>>>(A, lda, sAz, B, ldb, sBz, Cf, Cb, ldc, sCz, bias, cs, M, N, K, alpha, epi, causal);
}

static inline void cvt(hipStream_t st, const float* src, u16* dst, size_t n) {
    k_f32_to_bf16<<<(unsigned)((n + 255) / 256), 256, 0, st>>>(src, dst, (int)n);
}

extern "C" void kernel_launch(void* const* d_in, const int* in_sizes, int n_in,
                              void* d_out, int out_size, void* d_ws, size_t ws_size,
                              hipStream_t stream) {
    (void)in_sizes; (void)n_in; (void)out_size; (void)ws_size;
    const int S = S_TOK, C = CDIM;

    int ix = 0;
    const float* tokens       = (const float*)d_in[ix++];
    const float* token_norm_g = (const float*)d_in[ix++];
    const float* token_norm_b = (const float*)d_in[ix++];
    const float* trunk_norm_g = (const float*)d_in[ix++];
    const float* trunk_norm_b = (const float*)d_in[ix++];
    const float* empty_pose   = (const float*)d_in[ix++];
    const float* embed_w      = (const float*)d_in[ix++];
    const float* embed_b      = (const float*)d_in[ix++];
    const float* mod_w        = (const float*)d_in[ix++];
    const float* mod_b        = (const float*)d_in[ix++];
    const float* pb1_w        = (const float*)d_in[ix++];
    const float* pb1_b        = (const float*)d_in[ix++];
    const float* pb2_w        = (const float*)d_in[ix++];
    const float* pb2_b        = (const float*)d_in[ix++];
    struct Blk {
        const float *ln1_g, *ln1_b, *qkv_w, *qkv_b, *proj_w, *proj_b, *ls1;
        const float *ln2_g, *ln2_b, *fc1_w, *fc1_b, *fc2_w, *fc2_b, *ls2;
    } blk[4];
    for (int b = 0; b < 4; ++b) {
        blk[b].ln1_g  = (const float*)d_in[ix++]; blk[b].ln1_b  = (const float*)d_in[ix++];
        blk[b].qkv_w  = (const float*)d_in[ix++]; blk[b].qkv_b  = (const float*)d_in[ix++];
        blk[b].proj_w = (const float*)d_in[ix++]; blk[b].proj_b = (const float*)d_in[ix++];
        blk[b].ls1    = (const float*)d_in[ix++];
        blk[b].ln2_g  = (const float*)d_in[ix++]; blk[b].ln2_b  = (const float*)d_in[ix++];
        blk[b].fc1_w  = (const float*)d_in[ix++]; blk[b].fc1_b  = (const float*)d_in[ix++];
        blk[b].fc2_w  = (const float*)d_in[ix++]; blk[b].fc2_b  = (const float*)d_in[ix++];
        blk[b].ls2    = (const float*)d_in[ix++];
    }
    float* out = (float*)d_out;

    // ---- workspace bump allocator ----
    char* wp = (char*)d_ws;
    auto alloc = [&](size_t bytes) -> void* {
        void* p = (void*)wp;
        wp += (bytes + 255) & ~(size_t)255;
        return p;
    };
    u16 *wq[4], *wpj[4], *w1[4], *w2[4];
    for (int b = 0; b < 4; ++b) {
        wq[b]  = (u16*)alloc((size_t)C * C3 * 2);
        wpj[b] = (u16*)alloc((size_t)C * C * 2);
        w1[b]  = (u16*)alloc((size_t)C * HID * 2);
        w2[b]  = (u16*)alloc((size_t)HID * C * 2);
    }
    u16* wmod = (u16*)alloc((size_t)C * C3 * 2);
    u16* wpb1 = (u16*)alloc((size_t)C * 1024 * 2);

    float* pose  = (float*)alloc((size_t)S * C * 4);
    float* lnp   = (float*)alloc((size_t)S * C * 4);
    float* x     = (float*)alloc((size_t)S * C * 4);
    u16*   xb    = (u16*)  alloc((size_t)S * C * 2);
    u16*   msil  = (u16*)  alloc((size_t)S * C * 2);
    float* modb  = (float*)alloc((size_t)S * C3 * 4);
    u16*   qkv   = (u16*)  alloc((size_t)S * C3 * 2);
    float* scor  = (float*)alloc((size_t)HEADS * S * S * 4);
    u16*   probs = (u16*)  alloc((size_t)HEADS * S * S * 2);
    u16*   attno = (u16*)  alloc((size_t)S * C * 2);
    u16*   hmlp  = (u16*)  alloc((size_t)S * HID * 2);
    u16*   g1    = (u16*)  alloc((size_t)S * 1024 * 2);
    float* delta = (float*)alloc((size_t)S * 9 * 4);
    float* pred  = (float*)alloc((size_t)S * 9 * 4);

    // ---- one-time weight conversion fp32 -> bf16 ----
    for (int b = 0; b < 4; ++b) {
        cvt(stream, blk[b].qkv_w,  wq[b],  (size_t)C * C3);
        cvt(stream, blk[b].proj_w, wpj[b], (size_t)C * C);
        cvt(stream, blk[b].fc1_w,  w1[b],  (size_t)C * HID);
        cvt(stream, blk[b].fc2_w,  w2[b],  (size_t)HID * C);
    }
    cvt(stream, mod_w, wmod, (size_t)C * C3);
    cvt(stream, pb1_w, wpb1, (size_t)C * 1024);

    // ---- pose = LN(tokens[:,:,0], token_norm), lnp = LN(pose, eps=1e-6) ----
    k_layernorm<<<S, 256, 0, stream>>>(tokens, 2 * C, token_norm_g, token_norm_b, pose, nullptr, C, 1e-5f);
    k_layernorm<<<S, 256, 0, stream>>>(pose, C, nullptr, nullptr, lnp, nullptr, C, 1e-6f);

    const float qk_scale = 0.08838834764831845f;  // 1/sqrt(128)
    const long long SS = (long long)S * S;

    for (int it = 0; it < 4; ++it) {
        // modulation path
        k_embed_silu<<<dim3(C / 256, S), 256, 0, stream>>>(
            it == 0 ? empty_pose : pred, it == 0 ? 0 : 9, embed_w, embed_b, msil, C);
        gemm(stream, false, msil, C, 0, wmod, C3, 0, modb, nullptr, C3, 0,
             mod_b, nullptr, S, C3, C, 1.0f, EPI_F32, 0, 1);
        k_modulate<<<(S * C) / 256, 256, 0, stream>>>(pose, lnp, modb, x, C);

        for (int b = 0; b < 4; ++b) {
            // attention
            k_layernorm<<<S, 256, 0, stream>>>(x, C, blk[b].ln1_g, blk[b].ln1_b, nullptr, xb, C, 1e-5f);
            gemm(stream, false, xb, C, 0, wq[b], C3, 0, nullptr, qkv, C3, 0,
                 blk[b].qkv_b, nullptr, S, C3, C, 1.0f, EPI_BF16, 0, 1);
            // scores[h] = Q_h x K_h^T * scale  (causal tile skip)
            gemm(stream, true, qkv, C3, HDIM, qkv + C, C3, HDIM, scor, nullptr, S, SS,
                 nullptr, nullptr, S, S, HDIM, qk_scale, EPI_F32, 1, HEADS);
            k_softmax_causal<<<dim3(S, HEADS), 256, 0, stream>>>(scor, probs, S);
            // O_h = P_h x V_h   -> attno[:, h*128:(h+1)*128]
            gemm(stream, false, probs, S, SS, qkv + 2 * C, C3, HDIM, nullptr, attno, C, HDIM,
                 nullptr, nullptr, S, HDIM, S, 1.0f, EPI_BF16, 0, HEADS);
            // x += ls1 * (attno x proj_w + proj_b)
            gemm(stream, false, attno, C, 0, wpj[b], C, 0, x, nullptr, C, 0,
                 blk[b].proj_b, blk[b].ls1, S, C, C, 1.0f, EPI_F32_RES, 0, 1);
            // MLP
            k_layernorm<<<S, 256, 0, stream>>>(x, C, blk[b].ln2_g, blk[b].ln2_b, nullptr, xb, C, 1e-5f);
            gemm(stream, false, xb, C, 0, w1[b], HID, 0, nullptr, hmlp, HID, 0,
                 blk[b].fc1_b, nullptr, S, HID, C, 1.0f, EPI_BF16_GELU, 0, 1);
            gemm(stream, false, hmlp, HID, 0, w2[b], C, 0, x, nullptr, C, 0,
                 blk[b].fc2_b, blk[b].ls2, S, C, HID, 1.0f, EPI_F32_RES, 0, 1);
        }

        // head
        k_layernorm<<<S, 256, 0, stream>>>(x, C, trunk_norm_g, trunk_norm_b, nullptr, xb, C, 1e-5f);
        gemm(stream, false, xb, C, 0, wpb1, 1024, 0, nullptr, g1, 1024, 0,
             pb1_b, nullptr, S, 1024, C, 1.0f, EPI_BF16_GELU, 0, 1);
        k_pb2<<<S, 256, 0, stream>>>(g1, pb2_w, pb2_b, delta);
        k_pred_update<<<(S * 9 + 255) / 256, 256, 0, stream>>>(
            pred, delta, out + (size_t)it * S * 9, S * 9, it == 0 ? 1 : 0);
    }
}
